// _ModernBertAttention_22797686407734
// MI455X (gfx1250) — matrix operand order repaired and checked by reading, not compiled, not run on hardware
//
#include <hip/hip_runtime.h>
#include <hip/hip_bf16.h>

typedef __attribute__((ext_vector_type(16))) __bf16        v16bf;
typedef __attribute__((ext_vector_type(8)))  float         v8f;
typedef __attribute__((ext_vector_type(8)))  unsigned short us8;

#define BDIM  4
#define SDIM  2048
#define HDIM  1024
#define NHEAD 16
#define DHEAD 64

// ---------- small helpers ----------
__device__ __forceinline__ unsigned short f2bf(float f) {
  union { float f; unsigned u; } v; v.f = f;
  unsigned r = v.u + 0x7fffu + ((v.u >> 16) & 1u);   // round-to-nearest-even
  return (unsigned short)(r >> 16);
}
__device__ __forceinline__ float bf2f(unsigned short u) {
  union { unsigned u; float f; } v; v.u = ((unsigned)u) << 16; return v.f;
}

// A-fragment (16x32 bf16, M x K), rows at `base` with row stride ld (elements).
// CDNA5 layout: lanes0-15 M=lane, K = {0..7,16..23}; lanes16-31 M=lane-16, K={8..15,24..31}
__device__ __forceinline__ v16bf load_a_frag(const unsigned short* base, int ld, int lane) {
  const int half = lane >> 4;
  const int m    = lane & 15;
  const us8 c0 = *(const us8*)(base + m * ld + half * 8);
  const us8 c1 = *(const us8*)(base + m * ld + 16 + half * 8);
  v16bf a;
#pragma unroll
  for (int i = 0; i < 8; ++i) {
    a[i]     = __builtin_bit_cast(__bf16, (unsigned short)c0[i]);
    a[i + 8] = __builtin_bit_cast(__bf16, (unsigned short)c1[i]);
  }
  return a;
}

// B-fragment (32x16 bf16, K x N) from an LDS tile stored [n][k] with row stride ld.
// CDNA5 layout: lanes0-15 N=lane K={0..7,16..23}; lanes16-31 N=lane-16 K={8..15,24..31}
__device__ __forceinline__ v16bf load_b_frag(const unsigned short* base, int ld, int lane) {
  const int half = lane >> 4;
  const int n    = lane & 15;
  const us8 c0 = *(const us8*)(base + n * ld + half * 8);
  const us8 c1 = *(const us8*)(base + n * ld + 16 + half * 8);
  v16bf b;
#pragma unroll
  for (int i = 0; i < 8; ++i) {
    b[i]     = __builtin_bit_cast(__bf16, (unsigned short)c0[i]);
    b[i + 8] = __builtin_bit_cast(__bf16, (unsigned short)c1[i]);
  }
  return b;
}

__device__ __forceinline__ v8f wmma_bf16(v16bf a, v16bf b, v8f c) {
  return __builtin_amdgcn_wmma_f32_16x16x32_bf16(false, a, false, b, (short)0, c, false, false);
}

__device__ __forceinline__ float red_max16(float v) {
  v = fmaxf(v, __shfl_xor(v, 1));
  v = fmaxf(v, __shfl_xor(v, 2));
  v = fmaxf(v, __shfl_xor(v, 4));
  v = fmaxf(v, __shfl_xor(v, 8));
  return v;
}
__device__ __forceinline__ float red_sum16(float v) {
  v += __shfl_xor(v, 1);
  v += __shfl_xor(v, 2);
  v += __shfl_xor(v, 4);
  v += __shfl_xor(v, 8);
  return v;
}

// ---------- cast fp32 -> bf16 ----------
__global__ void cast_f32_bf16(const float* __restrict__ in, unsigned short* __restrict__ out,
                              size_t n) {
  size_t i = (size_t)blockIdx.x * blockDim.x + threadIdx.x;
  if (i < n) out[i] = f2bf(in[i]);
}

// ---------- generic bf16 NT GEMM: Y[m,n] = sum_k A[m,k]*B[n,k] ----------
// 128x128 block, 256 threads = 8 waves (2x4 wave grid), each wave 64x32 (4x2 frags).
// mode 0: fp32 out[m*N+n]
// mode 1: bf16 out[((b*NH+h)*S+s)*D+d]   (Q/K, head-major)
// mode 2: bf16 out[((b*NH+h)*D+d)*S+s]   (V transposed, for PV B-frags)
__global__ __launch_bounds__(256)
void gemm_bf16_nt(const unsigned short* __restrict__ A, const unsigned short* __restrict__ Bw,
                  void* __restrict__ out, int M, int N, int K, int mode) {
  __shared__ unsigned short lA[128 * 32];
  __shared__ unsigned short lB[128 * 32];

  const int tid  = threadIdx.x;
  const int lane = tid & 31;
  const int w    = tid >> 5;           // 0..7
  const int wm   = (w >> 2) * 64;      // 2 wave rows
  const int wn   = (w & 3) * 32;       // 4 wave cols
  const int m0   = blockIdx.y * 128;
  const int n0   = blockIdx.x * 128;

  v8f acc[4][2];
#pragma unroll
  for (int i = 0; i < 4; ++i)
#pragma unroll
    for (int j = 0; j < 2; ++j) acc[i][j] = (v8f){};

  for (int k0 = 0; k0 < K; k0 += 32) {
    __syncthreads();
    // 128 rows x 32 cols per tile = 512 us8 chunks per tile, 256 threads -> 2 each
#pragma unroll
    for (int c = tid; c < 512; c += 256) {
      const int r = c >> 2, cc = (c & 3) * 8;
      *(us8*)(lA + r * 32 + cc) = *(const us8*)(A  + (size_t)(m0 + r) * K + k0 + cc);
      *(us8*)(lB + r * 32 + cc) = *(const us8*)(Bw + (size_t)(n0 + r) * K + k0 + cc);
    }
    __syncthreads();

    v16bf af[4], bfv[2];
#pragma unroll
    for (int i = 0; i < 4; ++i) af[i] = load_a_frag(lA + (wm + i * 16) * 32, 32, lane);
#pragma unroll
    for (int j = 0; j < 2; ++j) bfv[j] = load_b_frag(lB + (wn + j * 16) * 32, 32, lane);

#pragma unroll
    for (int i = 0; i < 4; ++i)
#pragma unroll
      for (int j = 0; j < 2; ++j)
        acc[i][j] = wmma_bf16(af[i], bfv[j], acc[i][j]);
  }

  const int half = lane >> 4, nl = lane & 15;
#pragma unroll
  for (int i = 0; i < 4; ++i)
#pragma unroll
    for (int j = 0; j < 2; ++j)
#pragma unroll
      for (int r = 0; r < 8; ++r) {
        const int m = m0 + wm + i * 16 + r + half * 8;
        const int n = n0 + wn + j * 16 + nl;
        const float v = acc[i][j][r];
        if (mode == 0) {
          ((float*)out)[(size_t)m * N + n] = v;
        } else {
          const int b = m / SDIM, s = m % SDIM;
          const int h = n / DHEAD, d = n % DHEAD;
          const unsigned short bv = f2bf(v);
          if (mode == 1)
            ((unsigned short*)out)[(((size_t)b * NHEAD + h) * SDIM + s) * DHEAD + d] = bv;
          else
            ((unsigned short*)out)[(((size_t)b * NHEAD + h) * DHEAD + d) * SDIM + s] = bv;
        }
      }
}

// ---------- RoPE on Q and K, in place, layout [b,h,s,d] bf16 ----------
__global__ void rope_kernel(unsigned short* __restrict__ q, unsigned short* __restrict__ k,
                            const float* __restrict__ cosb, const float* __restrict__ sinb) {
  size_t i = (size_t)blockIdx.x * blockDim.x + threadIdx.x;
  if (i >= (size_t)BDIM * NHEAD * SDIM * (DHEAD / 2)) return;
  const int d   = (int)(i & 31);
  size_t rest   = i >> 5;
  const int s   = (int)(rest % SDIM);
  const size_t bh = rest / SDIM;
  const size_t base = (bh * SDIM + s) * DHEAD;

  const float c0 = cosb[(size_t)s * DHEAD + d];
  const float s0 = sinb[(size_t)s * DHEAD + d];
  const float c1 = cosb[(size_t)s * DHEAD + d + 32];
  const float s1 = sinb[(size_t)s * DHEAD + d + 32];

  {
    const float x0 = bf2f(q[base + d]), x1 = bf2f(q[base + d + 32]);
    q[base + d]      = f2bf(x0 * c0 - x1 * s0);
    q[base + d + 32] = f2bf(x1 * c1 + x0 * s1);
  }
  {
    const float x0 = bf2f(k[base + d]), x1 = bf2f(k[base + d + 32]);
    k[base + d]      = f2bf(x0 * c0 - x1 * s0);
    k[base + d + 32] = f2bf(x1 * c1 + x0 * s1);
  }
}

// ---------- flash attention: per (b,h, 64-query tile), 4 waves x 16 rows ----------
__global__ __launch_bounds__(128)
void attn_kernel(const unsigned short* __restrict__ Q, const unsigned short* __restrict__ K,
                 const unsigned short* __restrict__ Vt, const float* __restrict__ mask,
                 unsigned short* __restrict__ Ao) {
  __shared__ unsigned short lK[64 * 64];        // [key][d]
  __shared__ unsigned short lV[64 * 64];        // [d][key]
  __shared__ unsigned short lP[4][16 * 64];     // per-wave P staging [qrow][key]

  const int bh  = blockIdx.y;
  const int b   = bh / NHEAD;
  const int h   = bh % NHEAD;
  const int qt  = blockIdx.x;
  const int tid = threadIdx.x, lane = tid & 31, w = tid >> 5;
  const int half = lane >> 4, nl = lane & 15;
  const size_t qkbase = (size_t)bh * SDIM * DHEAD;
  const int qrow0 = qt * 64 + w * 16;
  const float scale = 0.125f;  // 1/sqrt(64)

  v16bf qa0, qa1;
  {
    const unsigned short* qp = Q + qkbase + (size_t)qrow0 * DHEAD;
    qa0 = load_a_frag(qp,      DHEAD, lane);   // d 0..31
    qa1 = load_a_frag(qp + 32, DHEAD, lane);   // d 32..63
  }

  v8f o[4];
#pragma unroll
  for (int t = 0; t < 4; ++t) o[t] = (v8f){};
  float rmax[8], rsum[8];
#pragma unroll
  for (int r = 0; r < 8; ++r) { rmax[r] = -1e30f; rsum[r] = 0.f; }

  for (int j0 = 0; j0 < SDIM; j0 += 64) {
    __syncthreads();
#pragma unroll
    for (int c = tid; c < 512; c += 128) {
      const int r = c >> 3, cc = (c & 7) * 8;
      *(us8*)(lK + r * 64 + cc) = *(const us8*)(K  + qkbase + (size_t)(j0 + r) * DHEAD + cc);
      *(us8*)(lV + r * 64 + cc) = *(const us8*)(Vt + qkbase + (size_t)r * SDIM + j0 + cc);
    }
    __syncthreads();

    // scores: 4 tiles of 16 keys
    float sc[4][8];
#pragma unroll
    for (int n16 = 0; n16 < 4; ++n16) {
      v16bf kb0 = load_b_frag(lK + (n16 * 16) * 64,      64, lane);  // d 0..31
      v16bf kb1 = load_b_frag(lK + (n16 * 16) * 64 + 32, 64, lane);  // d 32..63
      v8f c = (v8f){};
      c = wmma_bf16(qa0, kb0, c);
      c = wmma_bf16(qa1, kb1, c);
      const float mv = mask[(size_t)b * SDIM + j0 + n16 * 16 + nl];
#pragma unroll
      for (int r = 0; r < 8; ++r) sc[n16][r] = c[r] * scale + mv;
    }

    // online softmax
    float corr[8];
#pragma unroll
    for (int r = 0; r < 8; ++r) {
      float t = fmaxf(fmaxf(sc[0][r], sc[1][r]), fmaxf(sc[2][r], sc[3][r]));
      t = red_max16(t);
      const float nm = fmaxf(rmax[r], t);
      corr[r] = __expf(rmax[r] - nm);
      rmax[r] = nm;
    }
    float psum[8];
#pragma unroll
    for (int r = 0; r < 8; ++r) psum[r] = 0.f;
#pragma unroll
    for (int n16 = 0; n16 < 4; ++n16)
#pragma unroll
      for (int r = 0; r < 8; ++r) {
        const float p = __expf(sc[n16][r] - rmax[r]);
        psum[r] += p;
        lP[w][(r + half * 8) * 64 + n16 * 16 + nl] = f2bf(p);
      }
#pragma unroll
    for (int r = 0; r < 8; ++r) {
      rsum[r] = rsum[r] * corr[r] + red_sum16(psum[r]);
#pragma unroll
      for (int t = 0; t < 4; ++t) o[t][r] *= corr[r];
    }

    // PV: A = P (16 x 64 keys), B = V (keys x d)
    v16bf pa0 = load_a_frag(lP[w],      64, lane);   // keys 0..31
    v16bf pa1 = load_a_frag(lP[w] + 32, 64, lane);   // keys 32..63
#pragma unroll
    for (int d16 = 0; d16 < 4; ++d16) {
      v16bf vb0 = load_b_frag(lV + (d16 * 16) * 64,      64, lane);  // keys 0..31
      v16bf vb1 = load_b_frag(lV + (d16 * 16) * 64 + 32, 64, lane);  // keys 32..63
      o[d16] = wmma_bf16(pa0, vb0, o[d16]);
      o[d16] = wmma_bf16(pa1, vb1, o[d16]);
    }
  }

  // epilogue: normalize, write bf16 [B*S][H]
#pragma unroll
  for (int d16 = 0; d16 < 4; ++d16)
#pragma unroll
    for (int r = 0; r < 8; ++r) {
      const float v = o[d16][r] / rsum[r];
      const size_t row = (size_t)b * SDIM + qrow0 + r + half * 8;
      const int col = h * DHEAD + d16 * 16 + nl;
      Ao[row * HDIM + col] = f2bf(v);
    }
}

// ---------- host launcher ----------
extern "C" void kernel_launch(void* const* d_in, const int* in_sizes, int n_in,
                              void* d_out, int out_size, void* d_ws, size_t ws_size,
                              hipStream_t stream) {
  const float* hidden = (const float*)d_in[0];
  const float* mask   = (const float*)d_in[1];
  const float* cosb   = (const float*)d_in[2];
  const float* sinb   = (const float*)d_in[3];
  const float* Wq     = (const float*)d_in[4];
  const float* Wk     = (const float*)d_in[5];
  const float* Wv     = (const float*)d_in[6];
  const float* Wo     = (const float*)d_in[7];
  float* out = (float*)d_out;

  const size_t MT = (size_t)BDIM * SDIM;       // 8192 rows
  const size_t XE = MT * HDIM;                 // 8.39M elements
  const size_t WE = (size_t)HDIM * HDIM;       // 1.05M elements

  unsigned short* Xb  = (unsigned short*)d_ws;
  unsigned short* Wqb = Xb  + XE;
  unsigned short* Wkb = Wqb + WE;
  unsigned short* Wvb = Wkb + WE;
  unsigned short* Wob = Wvb + WE;
  unsigned short* Qb  = Wob + WE;
  unsigned short* Kb  = Qb  + XE;
  unsigned short* Vtb = Kb  + XE;
  unsigned short* Aob = Vtb + XE;

  // casts
  cast_f32_bf16<<<(unsigned)((XE + 255) / 256), 256, 0, stream>>>(hidden, Xb, XE);
  cast_f32_bf16<<<(unsigned)((WE + 255) / 256), 256, 0, stream>>>(Wq, Wqb, WE);
  cast_f32_bf16<<<(unsigned)((WE + 255) / 256), 256, 0, stream>>>(Wk, Wkb, WE);
  cast_f32_bf16<<<(unsigned)((WE + 255) / 256), 256, 0, stream>>>(Wv, Wvb, WE);
  cast_f32_bf16<<<(unsigned)((WE + 255) / 256), 256, 0, stream>>>(Wo, Wob, WE);

  // QKV projections (128x128 blocks)
  dim3 gg(HDIM / 128, (unsigned)(MT / 128));
  gemm_bf16_nt<<<gg, 256, 0, stream>>>(Xb, Wqb, Qb,  (int)MT, HDIM, HDIM, 1);
  gemm_bf16_nt<<<gg, 256, 0, stream>>>(Xb, Wkb, Kb,  (int)MT, HDIM, HDIM, 1);
  gemm_bf16_nt<<<gg, 256, 0, stream>>>(Xb, Wvb, Vtb, (int)MT, HDIM, HDIM, 2);

  // RoPE
  const size_t nr = (size_t)BDIM * NHEAD * SDIM * (DHEAD / 2);
  rope_kernel<<<(unsigned)((nr + 255) / 256), 256, 0, stream>>>(Qb, Kb, cosb, sinb);

  // attention
  dim3 ga(SDIM / 64, BDIM * NHEAD);
  attn_kernel<<<ga, 128, 0, stream>>>(Qb, Kb, Vtb, mask, Aob);

  // output projection -> fp32
  gemm_bf16_nt<<<gg, 256, 0, stream>>>(Aob, Wob, out, (int)MT, HDIM, HDIM, 0);
}
